// DSAVarlenSparseAttention_40372692583233
// MI455X (gfx1250) — matrix operand order repaired and checked by reading, not compiled, not run on hardware
//
#include <hip/hip_runtime.h>
#include <hip/hip_bf16.h>
#include <stdint.h>
#include <math.h>

// Problem constants (from reference): T=2048, H=8, D=128, DV=128, TOPK=64
#define T_TOK 2048
#define NH    8
#define HD    128
#define HDV   128
#define NTOPK 64

// LDS padding to kill bank conflicts (64 banks x 4B on CDNA5):
//  - K rows padded 128 -> 136 halfs (272B): A-frag b128 reads start at bank 4m+4hi
//    (optimal 2-way for 32xb128), gather b64 stores at banks 2*lane (conflict-free).
//  - Vt rows padded 64 -> 68 halfs (136B): transpose b16 stores at banks 34*lane
//    mod 64 (all distinct, conflict-free), frag reads on distinct groups.
#define KPAD  (HD + 8)     // 136
#define VPAD  (NTOPK + 4)  // 68

typedef __attribute__((ext_vector_type(16))) _Float16 v16h;
typedef __attribute__((ext_vector_type(8)))  float    v8f;

union Frag16 {
    v16h     v;
    uint32_t u[8];
};

// One wave (32 threads) handles one (t, h) pair.
//  Phase 1: gather 64 K rows (row-major f16) and 64 V rows (transposed f16) into LDS,
//           with global_prefetch 4 rows ahead (gather is L2-bandwidth bound).
//  Phase 2: S[64] = Kg[64x128] . q[128] via 16 x v_wmma_f32_16x16x32_f16
//           (q broadcast into all 16 B columns -> every column holds valid scores).
//  Phase 3: per-lane softmax over 32 scores + __shfl_xor(16) cross-half combine,
//           reweight by topk_scores, renormalize (matches reference exactly).
//  Phase 4: out[128] = w[64] . Vg[64x128] via 16 more WMMA (Vt as A, w broadcast as B).
__global__ __launch_bounds__(32)
void DSAVarlenSparseAttention_40372692583233_kernel(
    const float* __restrict__ q_packed,      // [T,H,D]  f32
    const float* __restrict__ k_packed,      // [T,H,D]  f32
    const float* __restrict__ v_packed,      // [T,H,DV] f32
    const long long* __restrict__ topk_idx,  // [T,K]    i64
    const float* __restrict__ topk_sc,       // [T,K]    f32
    float* __restrict__ out)                 // [T,H,DV] f32
{
    __shared__ _Float16 k_lds[NTOPK][KPAD];   // 17 KB, row-major (slot, d)
    __shared__ _Float16 vt_lds[HDV][VPAD];    // 17 KB, transposed (dv, slot)
    __shared__ _Float16 q_lds[HD];
    __shared__ _Float16 w_lds[NTOPK];
    __shared__ __align__(16) int idx_lds[NTOPK];
    __shared__ float    ts_lds[NTOPK];

    const int t    = blockIdx.x / NH;
    const int h    = blockIdx.x % NH;
    const int lane = threadIdx.x;            // 0..31, wave32
    const int m    = lane & 15;              // matrix row within tile
    const int hi   = lane >> 4;              // half-wave select

    const float* kbase = k_packed + (size_t)h * HD;
    const float* vbase = v_packed + (size_t)h * HDV;

    // ---- Phase 1: load indices/scores/q, gather K and V into LDS ------------
    {
        const long long i0 = topk_idx[(size_t)t * NTOPK + lane];
        const long long i1 = topk_idx[(size_t)t * NTOPK + 32 + lane];
        idx_lds[lane]      = (int)i0;
        idx_lds[32 + lane] = (int)i1;
        ts_lds[lane]       = topk_sc[(size_t)t * NTOPK + lane];
        ts_lds[32 + lane]  = topk_sc[(size_t)t * NTOPK + 32 + lane];

        const float4 qv =
            *(const float4*)(q_packed + ((size_t)t * NH + h) * HD + lane * 4);
        q_lds[lane * 4 + 0] = (_Float16)qv.x;
        q_lds[lane * 4 + 1] = (_Float16)qv.y;
        q_lds[lane * 4 + 2] = (_Float16)qv.z;
        q_lds[lane * 4 + 3] = (_Float16)qv.w;
    }
    __syncthreads();

    // 64 gathered rows, processed 4 at a time (one int4 index read per group).
    for (int rg = 0; rg < NTOPK; rg += 4) {
        const int4 id4 = *(const int4*)&idx_lds[rg];
        const int ids[4] = {id4.x, id4.y, id4.z, id4.w};
        #pragma unroll
        for (int ri = 0; ri < 4; ++ri) {
            const int r   = rg + ri;
            const int idx = ids[ri];

            // Prefetch the gather rows 4 iterations ahead (L2-resident K/V).
            if (r + 4 < NTOPK) {
                const int pidx = idx_lds[r + 4];
                __builtin_prefetch(kbase + (size_t)pidx * NH * HD + lane * 4, 0, 0);
                __builtin_prefetch(vbase + (size_t)pidx * NH * HDV + lane * 4, 0, 0);
            }

            // K row: lane moves one float4 (contiguous), store b64 -> banks 2*lane.
            const float4 kv =
                *(const float4*)(kbase + (size_t)idx * NH * HD + lane * 4);
            const int d = lane * 4;
            k_lds[r][d + 0] = (_Float16)kv.x;
            k_lds[r][d + 1] = (_Float16)kv.y;
            k_lds[r][d + 2] = (_Float16)kv.z;
            k_lds[r][d + 3] = (_Float16)kv.w;

            // V row: lane handles dv = lane + 32*i (4 coalesced 128B b32 loads),
            // transposed stores hit banks 34*lane mod 64 -> conflict-free.
            const float* vrow = vbase + (size_t)idx * NH * HDV;
            #pragma unroll
            for (int i = 0; i < 4; ++i) {
                const int dv = lane + 32 * i;
                vt_lds[dv][r] = (_Float16)vrow[dv];
            }
        }
    }
    __syncthreads();

    // ---- Phase 2: scores via WMMA ------------------------------------------
    // S[slot, n] = sum_d Kg[slot, d] * q[d]   (n broadcast, all columns equal)
    const float scale = 0.08838834764831845f; // 128^-0.5
    float s[4][8];
    #pragma unroll
    for (int mt = 0; mt < 4; ++mt) {         // 4 tiles of 16 topk slots
        v8f acc = {};
        #pragma unroll
        for (int kc = 0; kc < 4; ++kc) {     // contraction: 128 d in steps of 32
            Frag16 A, B;
            const int kb = kc * 32;
            #pragma unroll
            for (int vr = 0; vr < 8; ++vr) {
                // ISA 16-bit A layout: VGPR vr holds K = (vr>=4?16:0)+hi*8+2*(vr&3), +1
                const int ka = kb + ((vr >= 4) ? 16 : 0) + hi * 8 + (vr & 3) * 2;
                A.u[vr] = *(const uint32_t*)&k_lds[mt * 16 + m][ka];
                // B layout matches A: VGPR vr holds K = (vr>=4?16:0)+hi*8+2*(vr&3), +1;
                // 2 values per VGPR; q broadcast across all 16 columns.
                const int kq = kb + ((vr >= 4) ? 16 : 0) + hi * 8 + (vr & 3) * 2;
                B.u[vr] = *(const uint32_t*)&q_lds[kq];
            }
            acc = __builtin_amdgcn_wmma_f32_16x16x32_f16(
                false, A.v, false, B.v, (short)0, acc, false, false);
        }
        #pragma unroll
        for (int j = 0; j < 8; ++j) s[mt][j] = acc[j] * scale;
        // lane holds S rows { mt*16 + hi*8 + j }, identical across columns
    }

    // ---- Phase 3: softmax + topk_scores reweight + renormalize -------------
    float mx = -INFINITY;
    #pragma unroll
    for (int mt = 0; mt < 4; ++mt)
        #pragma unroll
        for (int j = 0; j < 8; ++j) mx = fmaxf(mx, s[mt][j]);
    mx = fmaxf(mx, __shfl_xor(mx, 16, 32));

    float e[4][8];
    float Z = 0.0f;
    #pragma unroll
    for (int mt = 0; mt < 4; ++mt)
        #pragma unroll
        for (int j = 0; j < 8; ++j) {
            e[mt][j] = __expf(s[mt][j] - mx);
            Z += e[mt][j];
        }
    Z += __shfl_xor(Z, 16, 32);
    const float invZ = 1.0f / Z;

    float U = 0.0f;
    float u[4][8];
    #pragma unroll
    for (int mt = 0; mt < 4; ++mt)
        #pragma unroll
        for (int j = 0; j < 8; ++j) {
            const int slot = mt * 16 + hi * 8 + j;
            u[mt][j] = e[mt][j] * invZ * ts_lds[slot];
            U += u[mt][j];
        }
    U += __shfl_xor(U, 16, 32);
    const float invU = 1.0f / (U + 1e-12f);

    if (m == 0) {   // one lane per half-wave writes w (values identical per column)
        #pragma unroll
        for (int mt = 0; mt < 4; ++mt)
            #pragma unroll
            for (int j = 0; j < 8; ++j)
                w_lds[mt * 16 + hi * 8 + j] = (_Float16)(u[mt][j] * invU);
    }
    __syncthreads();

    // ---- Phase 4: out = w . Vg via WMMA ------------------------------------
    // D[dv, n] = sum_slot Vt[dv, slot] * w[slot]   (n broadcast)
    #pragma unroll
    for (int dt = 0; dt < 8; ++dt) {         // 8 tiles of 16 dv
        v8f acc = {};
        #pragma unroll
        for (int kc = 0; kc < 2; ++kc) {     // contraction: 64 slots in steps of 32
            Frag16 A, B;
            const int kb = kc * 32;
            #pragma unroll
            for (int vr = 0; vr < 8; ++vr) {
                const int ka = kb + ((vr >= 4) ? 16 : 0) + hi * 8 + (vr & 3) * 2;
                A.u[vr] = *(const uint32_t*)&vt_lds[dt * 16 + m][ka];
                const int kw = kb + ((vr >= 4) ? 16 : 0) + hi * 8 + (vr & 3) * 2;
                B.u[vr] = *(const uint32_t*)&w_lds[kw];
            }
            acc = __builtin_amdgcn_wmma_f32_16x16x32_f16(
                false, A.v, false, B.v, (short)0, acc, false, false);
        }
        if (m == 0) {  // all 16 columns identical; lane 0 / lane 16 write their rows
            float* op = out + ((size_t)t * NH + h) * HDV + dt * 16 + hi * 8;
            #pragma unroll
            for (int j = 0; j < 8; ++j) op[j] = acc[j];
        }
    }
}

extern "C" void kernel_launch(void* const* d_in, const int* in_sizes, int n_in,
                              void* d_out, int out_size, void* d_ws, size_t ws_size,
                              hipStream_t stream) {
    const float*     q   = (const float*)d_in[0];
    const float*     k   = (const float*)d_in[1];
    const float*     v   = (const float*)d_in[2];
    const long long* idx = (const long long*)d_in[3];
    const float*     ts  = (const float*)d_in[4];
    float*           out = (float*)d_out;

    dim3 grid(T_TOK * NH);   // one wave32 workgroup per (t, h)
    dim3 block(32);
    hipLaunchKernelGGL(DSAVarlenSparseAttention_40372692583233_kernel,
                       grid, block, 0, stream, q, k, v, idx, ts, out);
}
